// SubwordSegmentalLM_32203664785472
// MI455X (gfx1250) — matrix operand order repaired and checked by reading, not compiled, not run on hardware
//
#include <hip/hip_runtime.h>
#include <math.h>

typedef _Float16 half_t;
typedef __attribute__((ext_vector_type(16))) _Float16 v16h;
typedef __attribute__((ext_vector_type(8)))  float    v8f;

#define DEVI static __device__ __forceinline__

constexpr int Bq = 16, Tq = 512, Lq = 8, Eq = 512, Hq = 1024, VCq = 256, VLq = 16384;
constexpr int BT = Bq * Tq;                 // 8192 rows
constexpr float NEGC = -1.0e30f;

DEVI float sigf(float x)  { return 1.0f / (1.0f + __expf(-x)); }
DEVI float lsigf(float x) { return fminf(x, 0.0f) - log1pf(__expf(-fabsf(x))); }

// ---------------- WMMA helpers (per CDNA5 ISA 7.12.2 layouts) ----------------
DEVI v8f wmma_f16(v16h a, v16h b, v8f c) {
  return __builtin_amdgcn_wmma_f32_16x16x32_f16(false, a, false, b, (short)0, c,
                                                false, false);
}

// A fragment: 16x32 f16 tile from row-major A (row stride lda elements).
// lanes 0-15: M=lane, K = k0+0..7 and k0+16..23 ; lanes 16-31: M=lane-16, K +8.
DEVI v16h frag_a(const half_t* A, int lda, int k0) {
  int lane = threadIdx.x & 31;
  const half_t* p = A + (size_t)(lane & 15) * lda + k0 + ((lane >> 4) << 3);
  v16h f;
#pragma unroll
  for (int i = 0; i < 8; ++i) { f[i] = p[i]; f[i + 8] = p[i + 16]; }
  return f;
}

// B fragment from PRE-TRANSPOSED weights Bt (N x K row-major, row stride ldk).
// lanes 0-15: N=n0+lane, K=k0+0..7 and k0+16..23 ; lanes 16-31: N=n0+lane-16, K +8.
// two runs of 8 contiguous halfs per lane -> two global_load_b128 (vs 16 strided u16).
DEVI v16h frag_bt(const half_t* Bt, int ldk, int k0, int n0) {
  int lane = threadIdx.x & 31;
  const half_t* p = Bt + (size_t)(n0 + (lane & 15)) * ldk + k0 + ((lane >> 4) << 3);
  v16h f;
#pragma unroll
  for (int i = 0; i < 8; ++i) { f[i] = p[i]; f[i + 8] = p[i + 16]; }
  return f;
}
// C layout: VGPR r -> row r (lanes 0-15) / row r+8 (lanes 16-31); col = lane&15.

// ---------------- utility kernels ----------------
__global__ void fill_f32(float* p, float v, size_t n) {
  size_t i = (size_t)blockIdx.x * blockDim.x + threadIdx.x;
  size_t st = (size_t)gridDim.x * blockDim.x;
  for (; i < n; i += st) p[i] = v;
}
__global__ void fill_f16(half_t* p, size_t n) {
  size_t i = (size_t)blockIdx.x * blockDim.x + threadIdx.x;
  size_t st = (size_t)gridDim.x * blockDim.x;
  for (; i < n; i += st) p[i] = (half_t)0.0f;
}
__global__ void f32_to_f16(const float* s, half_t* d, size_t n) {
  size_t i = (size_t)blockIdx.x * blockDim.x + threadIdx.x;
  size_t st = (size_t)gridDim.x * blockDim.x;
  for (; i < n; i += st) d[i] = (half_t)s[i];
}
// src (K x N, f32, row-major) -> dst (N x K, f16, row-major); K,N multiples of 32
__global__ void transpose_to_f16(const float* src, half_t* dst, int K, int N) {
  __shared__ float tile[32][33];
  int kb = blockIdx.x << 5, nb = blockIdx.y << 5;
  int tx = threadIdx.x & 31, ty = threadIdx.x >> 5;      // 32x8
#pragma unroll
  for (int i = ty; i < 32; i += 8)
    tile[i][tx] = src[(size_t)(kb + i) * N + nb + tx];
  __syncthreads();
#pragma unroll
  for (int i = ty; i < 32; i += 8)
    dst[(size_t)(nb + i) * K + kb + tx] = (half_t)tile[tx][i];
}
__global__ void embed_rows(const int* x, const half_t* embW, half_t* out) {
  int row = blockIdx.x;
  int id = x[row];
  for (int e = threadIdx.x; e < Eq; e += blockDim.x)
    out[(size_t)row * Eq + e] = embW[(size_t)id * Eq + e];
}

// ---------------- encoder LSTM: one time step (M=16 batch tile) ----------------
// WihT: (4H x E), WhhT: (4H x H)
__global__ void enc_step(const half_t* emb, const half_t* WihT, const half_t* WhhT,
                         const float* bias, const half_t* h_in, half_t* h_out,
                         float* c_st, float* hist, int t) {
  int j0 = blockIdx.x << 4;                    // 16 H-columns per wave
  v8f zi = {}, zf = {}, zg = {}, zo = {};
  const half_t* A = emb + (size_t)t * Eq;      // row b at stride Tq*Eq
#pragma unroll 1
  for (int k0 = 0; k0 < Eq; k0 += 32) {
    v16h a = frag_a(A, Tq * Eq, k0);
    zi = wmma_f16(a, frag_bt(WihT, Eq, k0, j0),          zi);
    zf = wmma_f16(a, frag_bt(WihT, Eq, k0, j0 + Hq),     zf);
    zg = wmma_f16(a, frag_bt(WihT, Eq, k0, j0 + 2 * Hq), zg);
    zo = wmma_f16(a, frag_bt(WihT, Eq, k0, j0 + 3 * Hq), zo);
  }
#pragma unroll 1
  for (int k0 = 0; k0 < Hq; k0 += 32) {
    v16h a = frag_a(h_in, Hq, k0);
    zi = wmma_f16(a, frag_bt(WhhT, Hq, k0, j0),          zi);
    zf = wmma_f16(a, frag_bt(WhhT, Hq, k0, j0 + Hq),     zf);
    zg = wmma_f16(a, frag_bt(WhhT, Hq, k0, j0 + 2 * Hq), zg);
    zo = wmma_f16(a, frag_bt(WhhT, Hq, k0, j0 + 3 * Hq), zo);
  }
  int lane = threadIdx.x & 31;
  int col = j0 + (lane & 15);
#pragma unroll
  for (int r = 0; r < 8; ++r) {
    int b = r + ((lane >> 4) << 3);
    float iv = sigf(zi[r] + bias[col]);
    float fv = sigf(zf[r] + bias[col + Hq]);
    float gv = tanhf(zg[r] + bias[col + 2 * Hq]);
    float ov = sigf(zo[r] + bias[col + 3 * Hq]);
    float cp = c_st[b * Hq + col];
    float cn = fv * cp + iv * gv;
    float hn = ov * tanhf(cn);
    c_st[b * Hq + col] = cn;
    h_out[b * Hq + col] = (half_t)hn;
    if (t + 1 < Tq) hist[((size_t)b * Tq + t + 1) * Hq + col] = hn;  // shifted
  }
}

// ---------------- decoder LSTM: one step over all B*T rows ----------------
DEVI int dec_in_token(const int* x, int row, int l) {
  int b = row >> 9, t = row & (Tq - 1);
  int m = t + l; m = m > Tq ? Tq : m;                    // min(idx, T) into x_pad
  int xm = x[b * Tq + (m > 0 ? m - 1 : 0)];
  return m == 0 ? 0 : xm;
}
DEVI v16h frag_a_embed(const half_t* embW, const int* x, int m0, int l, int k0) {
  int lane = threadIdx.x & 31;
  int id = dec_in_token(x, m0 + (lane & 15), l);
  const half_t* p = embW + (size_t)id * Eq + k0 + ((lane >> 4) << 3);
  v16h f;
#pragma unroll
  for (int i = 0; i < 8; ++i) { f[i] = p[i]; f[i + 8] = p[i + 16]; }
  return f;
}
__global__ void dec_step(const int* x, const half_t* embW,
                         const half_t* WihT, const half_t* WhhT, const float* bias,
                         const half_t* h_in, half_t* h_out, float* c_st, int l) {
  int m0 = blockIdx.x << 4;
  int j0 = blockIdx.y << 4;
  v8f zi = {}, zf = {}, zg = {}, zo = {};
#pragma unroll 1
  for (int k0 = 0; k0 < Eq; k0 += 32) {
    v16h a = frag_a_embed(embW, x, m0, l, k0);           // gather input embedding
    zi = wmma_f16(a, frag_bt(WihT, Eq, k0, j0),          zi);
    zf = wmma_f16(a, frag_bt(WihT, Eq, k0, j0 + Hq),     zf);
    zg = wmma_f16(a, frag_bt(WihT, Eq, k0, j0 + 2 * Hq), zg);
    zo = wmma_f16(a, frag_bt(WihT, Eq, k0, j0 + 3 * Hq), zo);
  }
#pragma unroll 1
  for (int k0 = 0; k0 < Hq; k0 += 32) {
    v16h a = frag_a(h_in + (size_t)m0 * Hq, Hq, k0);
    zi = wmma_f16(a, frag_bt(WhhT, Hq, k0, j0),          zi);
    zf = wmma_f16(a, frag_bt(WhhT, Hq, k0, j0 + Hq),     zf);
    zg = wmma_f16(a, frag_bt(WhhT, Hq, k0, j0 + 2 * Hq), zg);
    zo = wmma_f16(a, frag_bt(WhhT, Hq, k0, j0 + 3 * Hq), zo);
  }
  int lane = threadIdx.x & 31;
  int col = j0 + (lane & 15);
#pragma unroll
  for (int r = 0; r < 8; ++r) {
    int row = m0 + r + ((lane >> 4) << 3);
    float iv = sigf(zi[r] + bias[col]);
    float fv = sigf(zf[r] + bias[col + Hq]);
    float gv = tanhf(zg[r] + bias[col + 2 * Hq]);
    float ov = sigf(zo[r] + bias[col + 3 * Hq]);
    float cp = c_st[(size_t)row * Hq + col];
    float cn = fv * cp + iv * gv;
    c_st[(size_t)row * Hq + col] = cn;
    h_out[(size_t)row * Hq + col] = (half_t)(ov * tanhf(cn));
  }
}

// ------ generic WMMA GEMM vs transposed weights, 16x64 tile/wave -------------
__global__ void gemm_bias(const half_t* A, const half_t* Bt, const float* bias,
                          float* C, half_t* Ch, int K, int lda, int ldc) {
  int m0 = blockIdx.x << 4;
  int n0 = blockIdx.y << 6;
  v8f acc[4] = {};
#pragma unroll 1
  for (int k0 = 0; k0 < K; k0 += 32) {
    if (k0 + 32 < K) __builtin_prefetch(Bt + (size_t)n0 * K + k0 + 32, 0, 3);
    v16h a = frag_a(A + (size_t)m0 * lda, lda, k0);
#pragma unroll
    for (int j = 0; j < 4; ++j)
      acc[j] = wmma_f16(a, frag_bt(Bt, K, k0, n0 + (j << 4)), acc[j]);
  }
  int lane = threadIdx.x & 31;
#pragma unroll
  for (int j = 0; j < 4; ++j) {
    int col = n0 + (j << 4) + (lane & 15);
    float bv = bias ? bias[col] : 0.0f;
#pragma unroll
    for (int r = 0; r < 8; ++r) {
      int row = m0 + r + ((lane >> 4) << 3);
      float v = acc[j][r] + bv;
      if (C)  C[(size_t)row * ldc + col] = v;
      if (Ch) Ch[(size_t)row * ldc + col] = (half_t)v;
    }
  }
}

// ------- decoder char log-softmax (VC=256) + target gather, wave/row ----------
__global__ void dec_lse_gather(const float* logits, const int* x, float* lp_tgt, int l) {
  int wave = threadIdx.x >> 5, lane = threadIdx.x & 31;
  int row = (blockIdx.x << 3) + wave;
  const float* lr = logits + (size_t)row * VCq;
  float v[8], m = -3.0e38f;
#pragma unroll
  for (int i = 0; i < 8; ++i) { v[i] = lr[lane + (i << 5)]; m = fmaxf(m, v[i]); }
#pragma unroll
  for (int d = 16; d >= 1; d >>= 1) m = fmaxf(m, __shfl_xor(m, d));
  float s = 0.0f;
#pragma unroll
  for (int i = 0; i < 8; ++i) s += __expf(v[i] - m);
#pragma unroll
  for (int d = 16; d >= 1; d >>= 1) s += __shfl_xor(s, d);
  if (lane == 0) {
    int b = row >> 9, t = row & (Tq - 1);
    int mm = t + l; if (mm > Tq - 1) mm = Tq - 1;       // min(idx, T-1)
    int tgt = x[b * Tq + mm];
    lp_tgt[(size_t)l * BT + row] = lr[tgt] - (m + __logf(s));
  }
}

// ------- length-model logits + log-softmax over L=8, wave/row ----------------
__global__ void len_logits(const float* histf, const float* lenW, const float* lenb,
                           float* len_lp) {
  int wave = threadIdx.x >> 5, lane = threadIdx.x & 31;
  int row = (blockIdx.x << 3) + wave;
  float acc[Lq] = {};
  for (int k = lane; k < Hq; k += 32) {
    float hv = histf[(size_t)row * Hq + k];
#pragma unroll
    for (int j = 0; j < Lq; ++j) acc[j] += hv * lenW[k * Lq + j];
  }
#pragma unroll
  for (int j = 0; j < Lq; ++j)
#pragma unroll
    for (int d = 16; d >= 1; d >>= 1) acc[j] += __shfl_xor(acc[j], d);
  if (lane == 0) {
    float m = -3.0e38f;
#pragma unroll
    for (int j = 0; j < Lq; ++j) { acc[j] += lenb[j]; m = fmaxf(m, acc[j]); }
    float s = 0.0f;
#pragma unroll
    for (int j = 0; j < Lq; ++j) s += __expf(acc[j] - m);
    float lse = m + __logf(s);
#pragma unroll
    for (int j = 0; j < Lq; ++j) len_lp[(size_t)row * Lq + j] = acc[j] - lse;
  }
}

// ------- lexicon: online log-sum-exp over VL=16384 cols via WMMA -------------
// WfT: (VL x H). grid (BT/16 row tiles, 16 col-chunks of 1024).
__global__ void lex_lse_part(const half_t* Qh, const half_t* WfT, const float* fb,
                             float* part) {
  int m0 = blockIdx.x << 4;
  int chunk = blockIdx.y;
  int lane = threadIdx.x & 31;
  float rmax[8], rsum[8];
#pragma unroll
  for (int r = 0; r < 8; ++r) { rmax[r] = -3.0e38f; rsum[r] = 0.0f; }
  int cbase = chunk << 10;
#pragma unroll 1
  for (int n0 = cbase; n0 < cbase + 1024; n0 += 64) {
    v8f acc[4] = {};
#pragma unroll 1
    for (int k0 = 0; k0 < Hq; k0 += 32) {
      v16h a = frag_a(Qh + (size_t)m0 * Hq, Hq, k0);
#pragma unroll
      for (int j = 0; j < 4; ++j)
        acc[j] = wmma_f16(a, frag_bt(WfT, Hq, k0, n0 + (j << 4)), acc[j]);
    }
#pragma unroll
    for (int j = 0; j < 4; ++j) {
      float bv = fb[n0 + (j << 4) + (lane & 15)];
#pragma unroll
      for (int r = 0; r < 8; ++r) {
        float v = acc[j][r] + bv;
        if (v > rmax[r]) { rsum[r] *= __expf(rmax[r] - v); rmax[r] = v; }
        rsum[r] += __expf(v - rmax[r]);
      }
    }
  }
#pragma unroll
  for (int r = 0; r < 8; ++r) {
#pragma unroll
    for (int m = 1; m < 16; m <<= 1) {
      float om = __shfl_xor(rmax[r], m), os = __shfl_xor(rsum[r], m);
      float nm = fmaxf(rmax[r], om);
      rsum[r] = rsum[r] * __expf(rmax[r] - nm) + os * __expf(om - nm);
      rmax[r] = nm;
    }
  }
  if ((lane & 15) == 0) {
#pragma unroll
    for (int r = 0; r < 8; ++r) {
      int row = m0 + r + ((lane >> 4) << 3);
      part[((size_t)row * 16 + chunk) * 2 + 0] = rmax[r];
      part[((size_t)row * 16 + chunk) * 2 + 1] = rsum[r];
    }
  }
}
__global__ void lse_combine(const float* part, float* lse) {
  int row = blockIdx.x * blockDim.x + threadIdx.x;
  float m = -3.0e38f, s = 0.0f;
  for (int c = 0; c < 16; ++c) {
    float pm = part[((size_t)row * 16 + c) * 2 + 0];
    float ps = part[((size_t)row * 16 + c) * 2 + 1];
    float nm = fmaxf(m, pm);
    s = s * __expf(m - nm) + ps * __expf(pm - nm);
    m = nm;
  }
  lse[row] = m + __logf(s);
}
// gather the 8 lexicon logits per row (contiguous rows of WfT, same f16
// precision as the LSE pass so lp = logit - lse stays consistent)
__global__ void lex_gather(const float* q, const half_t* WfT, const float* fb,
                           const int* lex_ids, const float* lse, float* lp_lex) {
  int wave = threadIdx.x >> 5, lane = threadIdx.x & 31;
  int row = (blockIdx.x << 3) + wave;
#pragma unroll 1
  for (int j = 0; j < Lq; ++j) {
    int id = lex_ids[(size_t)row * Lq + j];
    float acc = 0.0f;
    for (int k = lane; k < Hq; k += 32)
      acc += q[(size_t)row * Hq + k] * (float)WfT[(size_t)id * Hq + k];
#pragma unroll
    for (int d = 16; d >= 1; d >>= 1) acc += __shfl_xor(acc, d);
    if (lane == 0) lp_lex[(size_t)row * Lq + j] = acc + fb[id] - lse[row];
  }
}

// ------- fused 4-way gate GEMM + gh.glW partial reduction (deterministic) ----
__global__ void gate_partial(const half_t* histh, const float* histf,
                             const half_t* AiT, const half_t* AfT,
                             const half_t* AgT, const half_t* AoT,
                             const float* bi, const float* bf,
                             const float* bg, const float* bo,
                             const float* glW, float* gpart) {
  int m0 = blockIdx.x << 4;
  int j0 = blockIdx.y << 4;
  v8f ci = {}, cf_ = {}, cg = {}, co = {};
#pragma unroll 1
  for (int k0 = 0; k0 < Hq; k0 += 32) {
    v16h a = frag_a(histh + (size_t)m0 * Hq, Hq, k0);
    ci  = wmma_f16(a, frag_bt(AiT, Hq, k0, j0), ci);
    cf_ = wmma_f16(a, frag_bt(AfT, Hq, k0, j0), cf_);
    cg  = wmma_f16(a, frag_bt(AgT, Hq, k0, j0), cg);
    co  = wmma_f16(a, frag_bt(AoT, Hq, k0, j0), co);
  }
  int lane = threadIdx.x & 31;
  int col = j0 + (lane & 15);
  float w = glW[col];
  float part[8];
#pragma unroll
  for (int r = 0; r < 8; ++r) {
    int row = m0 + r + ((lane >> 4) << 3);
    float iv = sigf(ci[r] + bi[col]);
    float fv = sigf(cf_[r] + bf[col]);
    float gv = tanhf(cg[r] + bg[col]);
    float ov = sigf(co[r] + bo[col]);
    float hv = histf[(size_t)row * Hq + col];
    part[r] = ov * tanhf(fv * hv + iv * gv) * w;
  }
#pragma unroll
  for (int r = 0; r < 8; ++r)
#pragma unroll
    for (int m = 1; m < 16; m <<= 1) part[r] += __shfl_xor(part[r], m);
  if ((lane & 15) == 0) {
#pragma unroll
    for (int r = 0; r < 8; ++r) {
      int row = m0 + r + ((lane >> 4) << 3);
      gpart[(size_t)row * 64 + blockIdx.y] = part[r];
    }
  }
}
__global__ void glog_reduce(const float* gpart, float* glog) {
  int row = blockIdx.x * blockDim.x + threadIdx.x;
  float s = 0.0f;
  for (int j = 0; j < 64; ++j) s += gpart[(size_t)row * 64 + j];
  glog[row] = s;
}

// ------- segment score combine: logaddexp(gate+char, (1-gate)+lex) -----------
__global__ void seg_combine(const float* lp_tgt, const float* len_lp,
                            const float* lp_lex, const float* glog,
                            const float* glb, float* seg) {
  int row = blockIdx.x * blockDim.x + threadIdx.x;
  int t = row & (Tq - 1);
  float gl = glog[row] + glb[0];
  float lsp = lsigf(gl), lsn = lsigf(-gl);
  float csum = 0.0f;
#pragma unroll
  for (int l = 0; l < Lq; ++l) {
    csum += lp_tgt[(size_t)l * BT + row];                // cumulative char lp
    float a = lsp + csum + len_lp[(size_t)row * Lq + l];
    float c = lsn + lp_lex[(size_t)row * Lq + l];
    float m = fmaxf(a, c);
    float v = m + __logf(__expf(a - m) + __expf(c - m));
    seg[(size_t)row * Lq + l] = (t + l) < Tq ? v : NEGC;
  }
}

// ------- sequential DP over T with window L, single block --------------------
__global__ void dp_scan(const float* seg, float* alpha, float* out) {
  int tid = threadIdx.x;
  int b = tid >> 3, k = tid & 7;                        // 16 batches x 8 lags
  const int W = Tq + 1 + Lq;
  for (int i = tid; i < Bq * W; i += blockDim.x) alpha[i] = NEGC;
  __syncthreads();
  if (tid < Bq) alpha[tid * W + Lq] = 0.0f;
  __syncthreads();
  for (int t = 1; t <= Tq; ++t) {
    int src = t - Lq + k;                               // segment start index
    float sv = (src >= 0) ? seg[((size_t)b * Tq + src) * Lq + (Lq - 1 - k)] : NEGC;
    float w = alpha[b * W + t + k] + sv;
    float m = w;
#pragma unroll
    for (int d = 1; d < 8; d <<= 1) m = fmaxf(m, __shfl_xor(m, d));
    float e = __expf(w - m);
#pragma unroll
    for (int d = 1; d < 8; d <<= 1) e += __shfl_xor(e, d);
    if (k == 0) alpha[b * W + t + Lq] = m + __logf(e);
    __syncthreads();
  }
  if (tid < Bq) out[tid] = -alpha[tid * W + Tq + Lq] / (float)Tq;
}

// =============================== host side ===================================
extern "C" void kernel_launch(void* const* d_in, const int* in_sizes, int n_in,
                              void* d_out, int out_size, void* d_ws, size_t ws_size,
                              hipStream_t stream) {
  (void)in_sizes; (void)n_in; (void)out_size; (void)ws_size;
  const int*   x         = (const int*)  d_in[0];
  const int*   lex_ids   = (const int*)  d_in[1];
  const float* embedding = (const float*)d_in[2];
  const float* enc_Wih   = (const float*)d_in[3];
  const float* enc_Whh   = (const float*)d_in[4];
  const float* enc_b     = (const float*)d_in[5];
  const float* dec_tW    = (const float*)d_in[6];
  const float* dec_tb    = (const float*)d_in[7];
  const float* dec_Wih   = (const float*)d_in[8];
  const float* dec_Whh   = (const float*)d_in[9];
  const float* dec_b     = (const float*)d_in[10];
  const float* dec_fW    = (const float*)d_in[11];
  const float* dec_fb    = (const float*)d_in[12];
  const float* gAi = (const float*)d_in[13]; const float* gbi = (const float*)d_in[14];
  const float* gAf = (const float*)d_in[15]; const float* gbf = (const float*)d_in[16];
  const float* gAg = (const float*)d_in[17]; const float* gbg = (const float*)d_in[18];
  const float* gAo = (const float*)d_in[19]; const float* gbo = (const float*)d_in[20];
  const float* glW = (const float*)d_in[21]; const float* glb = (const float*)d_in[22];
  const float* lex_tW = (const float*)d_in[23]; const float* lex_tb = (const float*)d_in[24];
  const float* lex_fW = (const float*)d_in[25]; const float* lex_fb = (const float*)d_in[26];
  const float* len_W  = (const float*)d_in[27]; const float* len_b  = (const float*)d_in[28];
  float* out = (float*)d_out;

  // ---- workspace arena (~270 MB) ----
  char* base = (char*)d_ws;
  size_t off = 0;
  auto alloc = [&](size_t bytes) -> void* {
    void* p = base + off;
    off = (off + bytes + 255) & ~(size_t)255;
    return p;
  };
  half_t* embW_h = (half_t*)alloc((size_t)VCq * Eq * 2);      // row-major (A side)
  half_t* eWihT  = (half_t*)alloc((size_t)Eq * 4 * Hq * 2);   // (4H x E)
  half_t* eWhhT  = (half_t*)alloc((size_t)Hq * 4 * Hq * 2);   // (4H x H)
  half_t* dtWT   = (half_t*)alloc((size_t)Hq * Hq * 2);       // (H x H)
  half_t* dWihT  = (half_t*)alloc((size_t)Eq * 4 * Hq * 2);
  half_t* dWhhT  = (half_t*)alloc((size_t)Hq * 4 * Hq * 2);
  half_t* dfWT   = (half_t*)alloc((size_t)Hq * VCq * 2);      // (VC x H)
  half_t* gAiT   = (half_t*)alloc((size_t)Hq * Hq * 2);
  half_t* gAfT   = (half_t*)alloc((size_t)Hq * Hq * 2);
  half_t* gAgT   = (half_t*)alloc((size_t)Hq * Hq * 2);
  half_t* gAoT   = (half_t*)alloc((size_t)Hq * Hq * 2);
  half_t* ltWT   = (half_t*)alloc((size_t)Hq * Hq * 2);
  half_t* lfWT   = (half_t*)alloc((size_t)Hq * VLq * 2);      // (VL x H)
  half_t* emb_h  = (half_t*)alloc((size_t)BT * Eq * 2);
  half_t* hist_h = (half_t*)alloc((size_t)BT * Hq * 2);
  float*  hist_f = (float*) alloc((size_t)BT * Hq * 4);
  half_t* ench0  = (half_t*)alloc((size_t)Bq * Hq * 2);
  half_t* ench1  = (half_t*)alloc((size_t)Bq * Hq * 2);
  float*  encc   = (float*) alloc((size_t)Bq * Hq * 4);
  half_t* dech0  = (half_t*)alloc((size_t)BT * Hq * 2);
  half_t* dech1  = (half_t*)alloc((size_t)BT * Hq * 2);
  float*  decc   = (float*) alloc((size_t)BT * Hq * 4);
  float*  dlog   = (float*) alloc((size_t)BT * VCq * 4);
  float*  lp_tgt = (float*) alloc((size_t)Lq * BT * 4);
  float*  len_lp = (float*) alloc((size_t)BT * Lq * 4);
  float*  qf     = (float*) alloc((size_t)BT * Hq * 4);
  half_t* qh     = (half_t*)alloc((size_t)BT * Hq * 2);
  float*  lsep   = (float*) alloc((size_t)BT * 16 * 2 * 4);
  float*  lexlse = (float*) alloc((size_t)BT * 4);
  float*  lp_lex = (float*) alloc((size_t)BT * Lq * 4);
  float*  gpart  = (float*) alloc((size_t)BT * 64 * 4);
  float*  glog   = (float*) alloc((size_t)BT * 4);
  float*  seg    = (float*) alloc((size_t)BT * Lq * 4);
  float*  alpha  = (float*) alloc((size_t)Bq * (Tq + 1 + Lq) * 4);

  // embedding stays row-major (A operand); all B weights go in transposed
  f32_to_f16<<<dim3((unsigned)(((size_t)VCq * Eq + 255) / 256)), 256, 0, stream>>>(
      embedding, embW_h, (size_t)VCq * Eq);
  auto trans = [&](const float* s, half_t* d, int K, int N) {
    transpose_to_f16<<<dim3(K / 32, N / 32), 256, 0, stream>>>(s, d, K, N);
  };
  trans(enc_Wih, eWihT, Eq, 4 * Hq);
  trans(enc_Whh, eWhhT, Hq, 4 * Hq);
  trans(dec_tW,  dtWT,  Hq, Hq);
  trans(dec_Wih, dWihT, Eq, 4 * Hq);
  trans(dec_Whh, dWhhT, Hq, 4 * Hq);
  trans(dec_fW,  dfWT,  Hq, VCq);
  trans(gAi, gAiT, Hq, Hq);
  trans(gAf, gAfT, Hq, Hq);
  trans(gAg, gAgT, Hq, Hq);
  trans(gAo, gAoT, Hq, Hq);
  trans(lex_tW, ltWT, Hq, Hq);
  trans(lex_fW, lfWT, Hq, VLq);

  embed_rows<<<BT, 256, 0, stream>>>(x, embW_h, emb_h);
  fill_f32<<<1024, 256, 0, stream>>>(hist_f, 0.0f, (size_t)BT * Hq);
  fill_f32<<<64, 256, 0, stream>>>(encc, 0.0f, (size_t)Bq * Hq);
  fill_f16<<<64, 256, 0, stream>>>(ench0, (size_t)Bq * Hq);

  // encoder: 512 sequential steps (latency-bound; M=16 = one WMMA tile)
  for (int t = 0; t < Tq; ++t) {
    half_t* hin  = (t & 1) ? ench1 : ench0;
    half_t* hout = (t & 1) ? ench0 : ench1;
    enc_step<<<dim3(Hq / 16), dim3(32), 0, stream>>>(emb_h, eWihT, eWhhT, enc_b,
                                                     hin, hout, encc, hist_f, t);
  }
  f32_to_f16<<<dim3((unsigned)(((size_t)BT * Hq + 255) / 256)), 256, 0, stream>>>(
      hist_f, hist_h, (size_t)BT * Hq);

  // dec_h0 = hist @ dec_tW + dec_tb (f16 out only)
  gemm_bias<<<dim3(BT / 16, Hq / 64), 32, 0, stream>>>(hist_h, dtWT, dec_tb,
                                                       nullptr, dech0, Hq, Hq, Hq);
  fill_f32<<<1024, 256, 0, stream>>>(decc, 0.0f, (size_t)BT * Hq);

  // decoder: 8 steps over 8192 rows; per-step char log-softmax + gather
  for (int l = 0; l < Lq; ++l) {
    half_t* hin  = (l & 1) ? dech1 : dech0;
    half_t* hout = (l & 1) ? dech0 : dech1;
    dec_step<<<dim3(BT / 16, Hq / 16), 32, 0, stream>>>(x, embW_h, dWihT, dWhhT,
                                                        dec_b, hin, hout, decc, l);
    gemm_bias<<<dim3(BT / 16, VCq / 64), 32, 0, stream>>>(hout, dfWT, dec_fb,
                                                          dlog, nullptr, Hq, Hq, VCq);
    dec_lse_gather<<<BT / 8, 256, 0, stream>>>(dlog, x, lp_tgt, l);
  }

  len_logits<<<BT / 8, 256, 0, stream>>>(hist_f, len_W, len_b, len_lp);

  // lexicon path
  gemm_bias<<<dim3(BT / 16, Hq / 64), 32, 0, stream>>>(hist_h, ltWT, lex_tb,
                                                       qf, qh, Hq, Hq, Hq);
  lex_lse_part<<<dim3(BT / 16, 16), 32, 0, stream>>>(qh, lfWT, lex_fb, lsep);
  lse_combine<<<BT / 256, 256, 0, stream>>>(lsep, lexlse);
  lex_gather<<<BT / 8, 256, 0, stream>>>(qf, lfWT, lex_fb, lex_ids, lexlse, lp_lex);

  // gate MLP + glog
  gate_partial<<<dim3(BT / 16, Hq / 16), 32, 0, stream>>>(hist_h, hist_f,
      gAiT, gAfT, gAgT, gAoT, gbi, gbf, gbg, gbo, glW, gpart);
  glog_reduce<<<BT / 256, 256, 0, stream>>>(gpart, glog);

  seg_combine<<<BT / 256, 256, 0, stream>>>(lp_tgt, len_lp, lp_lex, glog, glb, seg);
  dp_scan<<<1, 128, 0, stream>>>(seg, alpha, out);
}
